// LoRACausalSelfAttention_52192442581221
// MI455X (gfx1250) — matrix operand order repaired and checked by reading, not compiled, not run on hardware
//
#include <hip/hip_runtime.h>
#include <hip/hip_bf16.h>
#include <stdint.h>

typedef __attribute__((ext_vector_type(16))) __bf16 v16bf;
typedef __attribute__((ext_vector_type(8)))  float  v8f;
typedef __attribute__((ext_vector_type(4)))  unsigned int v4u_t;
typedef __attribute__((ext_vector_type(8)))  int    v8i_t;
typedef __attribute__((ext_vector_type(4)))  int    v4i_t;

#define HID_ 1024
#define NH_  16
#define HD_  64
#define RR_  8
#define TT_  2048
#define BB_  4
#define NT_  (BB_*TT_)      // 8192 token rows
#define SCALING_ 2.0f

// workspace byte offsets (256-aligned)
#define OFF_XBF 0u
#define OFF_WBF 16777216u           // 3 x 1024x1024 bf16
#define OFF_LX  23068672u           // 3 x 8192 x 8 f32
#define OFF_Q   23855104u           // bf16 [B,H,T,D]
#define OFF_K   40632320u           // bf16 [B,H,T,D]
#define OFF_VT  57409536u           // bf16 [B,H,D,T] (transposed V)

#if __has_include(<hip/amd_detail/amd_gfx1250_TDM.h>)
#define TDM_6ARG 1
#endif
#if __has_builtin(__builtin_amdgcn_tensor_load_to_lds)
#define USE_TDM 1
#endif

union FragU { v16bf v; uint4 q[2]; };

__device__ __forceinline__ unsigned short f2bf(float f) {
  union { float f; unsigned u; } v; v.f = f;
  unsigned r = v.u + 0x7FFFu + ((v.u >> 16) & 1u);   // round-to-nearest-even
  return (unsigned short)(r >> 16);
}

__device__ __forceinline__ v8f wmma_bf16(v16bf a, v16bf b, v8f c) {
  return __builtin_amdgcn_wmma_f32_16x16x32_bf16(false, a, false, b, (short)0, c, false, false);
}

#ifdef USE_TDM
// TDM: load a 64(o) x 32(k) bf16 tile of W (row stride 1024 elems) into LDS.
// D# group0: count=1 | lds_addr | global_addr(57b) | type=2 (bits 127:126)
// D# group1: data_size=1(2B); tensor_dim0=1024; tensor_dim1=1024;
//            tile_dim0=32; tile_dim1=64; tensor_dim0_stride=1024
__device__ __forceinline__ void tdm_load_w_tile(const unsigned short* gptr,
                                                unsigned lds_off) {
  unsigned long long ga = (unsigned long long)(uintptr_t)gptr;
  v4u_t g0 = { 1u, lds_off,
               (unsigned)(ga & 0xffffffffull),
               (unsigned)((ga >> 32) & 0x1ffffffull) | 0x80000000u };
  v8i_t g1 = { 0x00010000,          // data_size = 1 (2 bytes)
               0x04000000,          // tensor_dim0 lo16 = 1024 in [31:16]
               0x04000000,          // tensor_dim1 lo16 = 1024 in [31:16]
               (int)(32u << 16),    // tile_dim0 = 32 in [31:16]
               64,                  // tile_dim1 = 64 in [15:0]
               1024,                // tensor_dim0_stride lo32
               0, 0 };
  v4i_t z4 = { 0, 0, 0, 0 };
#ifdef TDM_6ARG
  v8i_t z8 = { 0, 0, 0, 0, 0, 0, 0, 0 };
  __builtin_amdgcn_tensor_load_to_lds(g0, g1, z4, z4, z8, 0);
#else
  __builtin_amdgcn_tensor_load_to_lds(g0, g1, z4, z4, 0);
#endif
}
#endif

// ---------------- fp32 -> bf16 conversion ----------------
__global__ void cvt_bf16_kernel(const float* __restrict__ src,
                                unsigned short* __restrict__ dst, int n) {
  int i = blockIdx.x * blockDim.x + threadIdx.x;
  int stride = gridDim.x * blockDim.x;
  for (; i < n; i += stride) dst[i] = f2bf(src[i]);
}

// ---------------- LoRA rank-8: lx[p][t][r] = x[t,:] . A_p[r,:] ----------------
__global__ void lora_x_kernel(const float* __restrict__ x,
                              const float* __restrict__ Aq,
                              const float* __restrict__ Ak,
                              const float* __restrict__ Av,
                              float* __restrict__ lx) {
  int lane = threadIdx.x & 31;
  int grp  = threadIdx.x >> 5;
  int t = blockIdx.x * 8 + grp;
  if (t >= NT_ || lane >= 24) return;
  int p = lane >> 3, r = lane & 7;
  const float* A  = (p == 0) ? Aq : (p == 1) ? Ak : Av;
  const float* xr = x + (size_t)t * HID_;
  const float* Ar = A + (size_t)r * HID_;
  float s = 0.f;
  for (int i = 0; i < HID_; i += 4)
    s += xr[i]*Ar[i] + xr[i+1]*Ar[i+1] + xr[i+2]*Ar[i+2] + xr[i+3]*Ar[i+3];
  lx[((size_t)p * NT_ + t) * RR_ + r] = s;
}

// ---------------- QKV GEMM ----------------
// Block = 8 waves sharing one (p, n0=64-col tile); wave w owns 32 token rows.
// W k-chunk tile (64x32 bf16, 4KB) staged into double-buffered LDS by the
// Tensor Data Mover (wave 0 issues, s_wait_tensorcnt, block barriers).
__global__ __launch_bounds__(256) void qkv_gemm_kernel(
    const unsigned short* __restrict__ xbf,
    const unsigned short* __restrict__ wbf,
    const float* __restrict__ lx,
    const float* __restrict__ bq, const float* __restrict__ bk, const float* __restrict__ bv,
    const float* __restrict__ Bq, const float* __restrict__ Bk, const float* __restrict__ Bv,
    unsigned short* __restrict__ Qb, unsigned short* __restrict__ Kb,
    unsigned short* __restrict__ Vt)
{
  __shared__ __align__(16) unsigned short Wlds[2][64 * 32];   // 2 x 4KB

  int wid  = threadIdx.x >> 5;
  int lane = threadIdx.x & 31;
  int lm = lane & 15, g = lane >> 4;

  // grid: 3 proj x 16 ntiles x 32 mblocks = 1536 blocks
  int p      = blockIdx.x / 512;
  int rem    = blockIdx.x % 512;
  int n0     = (rem >> 5) * 64;
  int m0     = (rem & 31) * 256 + wid * 32;

  const unsigned short* wtile = wbf + (size_t)p * HID_ * HID_ + (size_t)n0 * HID_;
  const uint4* arow0 = (const uint4*)(xbf + (size_t)(m0 +      lm) * HID_);
  const uint4* arow1 = (const uint4*)(xbf + (size_t)(m0 + 16 + lm) * HID_);

  v8f zero = {0.f,0.f,0.f,0.f,0.f,0.f,0.f,0.f};
  v8f acc[2][4] = {{zero,zero,zero,zero},{zero,zero,zero,zero}};

  // stage chunk 0 into buffer 0
#ifdef USE_TDM
  if (wid == 0)
    tdm_load_w_tile(wtile, (unsigned)(uintptr_t)&Wlds[0][0]);
#else
  {
    int tt = threadIdx.x;                       // 256 threads x 16B = 4KB
    int row = tt >> 2, col = (tt & 3) * 8;
    uint4 v = *(const uint4*)(wtile + (size_t)row * HID_ + col);
    *(uint4*)&Wlds[0][row * 32 + col] = v;
  }
#endif

  int buf = 0;
  for (int kc = 0; kc < HID_; kc += 32) {
    int more = (kc + 32) < HID_;
    // stage next chunk into the other buffer (freed by previous iteration's barrier)
#ifdef USE_TDM
    if (wid == 0) {
      if (more)
        tdm_load_w_tile(wtile + kc + 32, (unsigned)(uintptr_t)&Wlds[buf ^ 1][0]);
      if (more) __builtin_amdgcn_s_wait_tensorcnt(1);
      else      __builtin_amdgcn_s_wait_tensorcnt(0);
    }
#else
    if (more) {
      int tt = threadIdx.x;
      int row = tt >> 2, col = (tt & 3) * 8;
      uint4 v = *(const uint4*)(wtile + (size_t)row * HID_ + kc + 32 + col);
      *(uint4*)&Wlds[buf ^ 1][row * 32 + col] = v;
    }
#endif
    __syncthreads();   // chunk kc visible in Wlds[buf] to all waves

    FragU a0, a1;
    a0.q[0] = arow0[(kc +      8*g) >> 3];
    a0.q[1] = arow0[(kc + 16 + 8*g) >> 3];
    a1.q[0] = arow1[(kc +      8*g) >> 3];
    a1.q[1] = arow1[(kc + 16 + 8*g) >> 3];
#pragma unroll
    for (int j = 0; j < 4; ++j) {
      const uint4* bp = (const uint4*)&Wlds[buf][(j*16 + lm) * 32];
      FragU bf;
      bf.q[0] = bp[g];
      bf.q[1] = bp[2 + g];
      acc[0][j] = wmma_bf16(a0.v, bf.v, acc[0][j]);
      acc[1][j] = wmma_bf16(a1.v, bf.v, acc[1][j]);
    }
    __syncthreads();   // all done reading Wlds[buf] before it is re-staged
    buf ^= 1;
  }

  // ---- epilogue: + bias + 2 * (lx . B^T), write bf16 Q/K [B,H,T,D], V^T [B,H,D,T] ----
  const float* bias = (p==0) ? bq : (p==1) ? bk : bv;
  const float* Bm   = (p==0) ? Bq : (p==1) ? Bk : Bv;
  const float* lxp  = lx + (size_t)p * NT_ * RR_;

  float  bo[4];
  float4 c0[4], c1[4];
  int    hh[4], dd[4];
#pragma unroll
  for (int j = 0; j < 4; ++j) {
    int o = n0 + j*16 + lm;
    bo[j] = bias[o];
    const float4* B4 = (const float4*)(Bm + (size_t)o * RR_);
    c0[j] = B4[0]; c1[j] = B4[1];
    hh[j] = o >> 6; dd[j] = o & 63;
  }

#pragma unroll
  for (int mh = 0; mh < 2; ++mh) {
#pragma unroll
    for (int r = 0; r < 8; ++r) {
      int t = m0 + mh*16 + r + 8*g;
      const float4* l4 = (const float4*)(lxp + (size_t)t * RR_);
      float4 x0 = l4[0], x1 = l4[1];
      int b = t >> 11, tl = t & (TT_ - 1);
#pragma unroll
      for (int j = 0; j < 4; ++j) {
        float lor = x0.x*c0[j].x + x0.y*c0[j].y + x0.z*c0[j].z + x0.w*c0[j].w
                  + x1.x*c1[j].x + x1.y*c1[j].y + x1.z*c1[j].z + x1.w*c1[j].w;
        float val = acc[mh][j][r] + bo[j] + SCALING_ * lor;
        unsigned short hv = f2bf(val);
        if (p == 0)      Qb[(((size_t)b*NH_ + hh[j])*TT_ + tl)*HD_ + dd[j]] = hv;
        else if (p == 1) Kb[(((size_t)b*NH_ + hh[j])*TT_ + tl)*HD_ + dd[j]] = hv;
        else             Vt[(((size_t)b*NH_ + hh[j])*HD_ + dd[j])*TT_ + tl] = hv;
      }
    }
  }
}

// ---------------- causal flash attention: one wave per (b,h,16-row q tile) ----------------
__global__ __launch_bounds__(256) void attn_kernel(
    const unsigned short* __restrict__ Qb,
    const unsigned short* __restrict__ Kb,
    const unsigned short* __restrict__ Vt,
    const float* __restrict__ amask,
    float* __restrict__ out)
{
  __shared__ __align__(16) unsigned short pbuf[8][16*32]; // per-wave P staging (bf16)
  int wid  = threadIdx.x >> 5;
  int lane = threadIdx.x & 31;
  int wave = blockIdx.x * 8 + wid;          // 8192 waves total
  int qtile = wave & 127;
  int bh    = wave >> 7;                    // b*16 + h
  int b     = bh >> 4;
  int q0    = qtile * 16;
  int lm = lane & 15, g = lane >> 4;

  const uint4* q4 = (const uint4*)(Qb + ((size_t)bh * TT_ + q0 + lm) * HD_);
  FragU qf0, qf1;
  qf0.q[0] = q4[(16*g    ) >> 3]; qf0.q[1] = q4[(16*g + 8) >> 3];
  qf1.q[0] = q4[(32 + 16*g) >> 3]; qf1.q[1] = q4[(40 + 16*g) >> 3];

  float mi[8], li[8];
  v8f zero = {0.f,0.f,0.f,0.f,0.f,0.f,0.f,0.f};
  v8f acc[4] = {zero, zero, zero, zero};
#pragma unroll
  for (int r = 0; r < 8; ++r) { mi[r] = -1e30f; li[r] = 0.f; }

  const float* mrow = amask + (size_t)b * TT_;
  const unsigned short* Krows = Kb + (size_t)bh * TT_ * HD_;
  const unsigned short* Vrows = Vt + (size_t)bh * HD_ * TT_;

  for (int kc = 0; kc <= q0 + 15; kc += 32) {
    __builtin_prefetch(Krows + (size_t)(kc + 32 + lm) * HD_, 0, 1);
    __builtin_prefetch(Vrows + (size_t)lm * TT_ + kc + 32, 0, 1);

    // ---- scores S0 (keys kc..kc+15), S1 (keys kc+16..kc+31) ----
    v8f s0 = zero, s1 = zero;
    {
      const uint4* k4 = (const uint4*)(Krows + (size_t)(kc + lm) * HD_);
      FragU kf;
      kf.q[0] = k4[(16*g    ) >> 3]; kf.q[1] = k4[(16*g + 8) >> 3];
      s0 = wmma_bf16(qf0.v, kf.v, s0);
      kf.q[0] = k4[(32 + 16*g) >> 3]; kf.q[1] = k4[(40 + 16*g) >> 3];
      s0 = wmma_bf16(qf1.v, kf.v, s0);
      const uint4* k4b = (const uint4*)(Krows + (size_t)(kc + 16 + lm) * HD_);
      kf.q[0] = k4b[(16*g    ) >> 3]; kf.q[1] = k4b[(16*g + 8) >> 3];
      s1 = wmma_bf16(qf0.v, kf.v, s1);
      kf.q[0] = k4b[(32 + 16*g) >> 3]; kf.q[1] = k4b[(40 + 16*g) >> 3];
      s1 = wmma_bf16(qf1.v, kf.v, s1);
    }

    float mk0 = mrow[kc + lm];
    float mk1 = mrow[kc + 16 + lm];

    // ---- online softmax (row stats across 16 N-lanes via shfl_xor) ----
#pragma unroll
    for (int r = 0; r < 8; ++r) {
      int qrow = q0 + r + 8*g;
      float a0 = s0[r] * 0.125f + mk0;
      float a1 = s1[r] * 0.125f + mk1;
      if (kc + lm      > qrow) a0 = -1e30f;
      if (kc + 16 + lm > qrow) a1 = -1e30f;

      float mx = fmaxf(a0, a1);
#pragma unroll
      for (int off = 8; off >= 1; off >>= 1)
        mx = fmaxf(mx, __shfl_xor(mx, off, 32));
      float nm    = fmaxf(mi[r], mx);
      float alpha = __expf(mi[r] - nm);
      float e0 = __expf(a0 - nm);
      float e1 = __expf(a1 - nm);
      float rs = e0 + e1;
#pragma unroll
      for (int off = 8; off >= 1; off >>= 1)
        rs += __shfl_xor(rs, off, 32);
      li[r] = li[r] * alpha + rs;
      mi[r] = nm;
#pragma unroll
      for (int dt = 0; dt < 4; ++dt) acc[dt][r] *= alpha;

      int m = r + 8*g;                       // C-layout -> LDS row-major P[16][32]
      pbuf[wid][m*32 + lm]      = f2bf(e0);
      pbuf[wid][m*32 + 16 + lm] = f2bf(e1);
    }

    // wave-private LDS: make stores visible to cross-lane reads
    asm volatile("s_wait_dscnt 0" ::: "memory");

    FragU pf;
    const uint4* pl = (const uint4*)&pbuf[wid][lm * 32];
    pf.q[0] = pl[2*g];
    pf.q[1] = pl[2*g + 1];

    // ---- O += P . V (V pre-transposed: contiguous keys per d row) ----
#pragma unroll
    for (int dt = 0; dt < 4; ++dt) {
      const uint4* v4 = (const uint4*)(Vrows + (size_t)(dt*16 + lm) * TT_ + kc);
      FragU vf;
      vf.q[0] = v4[(16*g    ) >> 3];
      vf.q[1] = v4[(16*g + 8) >> 3];
      acc[dt] = wmma_bf16(pf.v, vf.v, acc[dt]);
    }
  }

  // ---- normalize and write out [B,T,H*D] fp32 ----
  int h = bh & 15;
#pragma unroll
  for (int r = 0; r < 8; ++r) {
    float inv = 1.0f / li[r];
    int t = q0 + r + 8*g;
    float* orow = out + ((size_t)b * TT_ + t) * HID_ + h * HD_;
#pragma unroll
    for (int dt = 0; dt < 4; ++dt)
      orow[dt*16 + lm] = acc[dt][r] * inv;
  }
}

extern "C" void kernel_launch(void* const* d_in, const int* in_sizes, int n_in,
                              void* d_out, int out_size, void* d_ws, size_t ws_size,
                              hipStream_t stream) {
  (void)in_sizes; (void)n_in; (void)out_size; (void)ws_size;
  const float* x  = (const float*)d_in[0];
  const float* am = (const float*)d_in[1];
  const float* Wq = (const float*)d_in[2];
  const float* bq = (const float*)d_in[3];
  const float* Aq = (const float*)d_in[4];
  const float* Bq = (const float*)d_in[5];
  const float* Wk = (const float*)d_in[6];
  const float* bk = (const float*)d_in[7];
  const float* Ak = (const float*)d_in[8];
  const float* Bk = (const float*)d_in[9];
  const float* Wv = (const float*)d_in[10];
  const float* bv = (const float*)d_in[11];
  const float* Av = (const float*)d_in[12];
  const float* Bv = (const float*)d_in[13];

  char* ws = (char*)d_ws;
  unsigned short* xbf = (unsigned short*)(ws + OFF_XBF);
  unsigned short* wbf = (unsigned short*)(ws + OFF_WBF);
  float*          lx  = (float*)        (ws + OFF_LX);
  unsigned short* Qb  = (unsigned short*)(ws + OFF_Q);
  unsigned short* Kb  = (unsigned short*)(ws + OFF_K);
  unsigned short* Vt  = (unsigned short*)(ws + OFF_VT);

  cvt_bf16_kernel<<<2048, 256, 0, stream>>>(x,  xbf,               NT_ * HID_);
  cvt_bf16_kernel<<<1024, 256, 0, stream>>>(Wq, wbf,               HID_ * HID_);
  cvt_bf16_kernel<<<1024, 256, 0, stream>>>(Wk, wbf + 1*HID_*HID_, HID_ * HID_);
  cvt_bf16_kernel<<<1024, 256, 0, stream>>>(Wv, wbf + 2*HID_*HID_, HID_ * HID_);

  lora_x_kernel<<<NT_ / 8, 256, 0, stream>>>(x, Aq, Ak, Av, lx);

  // 3 proj x 16 n-tiles x 32 m-blocks = 1536 blocks (8 waves x 32 rows each)
  qkv_gemm_kernel<<<1536, 256, 0, stream>>>(xbf, wbf, lx,
                                            bq, bk, bv, Bq, Bk, Bv,
                                            Qb, Kb, Vt);

  // 4*16*128 = 8192 waves / 8 per block
  attn_kernel<<<1024, 256, 0, stream>>>(Qb, Kb, Vt, am, (float*)d_out);
}
